// LinearAttention_71287867179226
// MI455X (gfx1250) — compile-verified, matrix operand order checked by reading, not run on hardware
//
#include <hip/hip_runtime.h>
#include <hip/hip_bf16.h>

typedef __bf16 bf16_t;
typedef __attribute__((ext_vector_type(16))) __bf16 v16bf;
typedef __attribute__((ext_vector_type(8)))  float  v8f;

#define HNUM  8
#define EDIM  64
#define CDIM  512
#define NSEQ  4096
#define BATCH 16
#define MROWS (BATCH * NSEQ)   /* 65536 */
#define QKVN  (3 * CDIM)       /* 1536  */

__device__ __forceinline__ int lane_id() { return threadIdx.x & 31; }

// Async global->LDS copy (CDNA5 ASYNCcnt path). ISA 10.2: generic LDS pointer
// low 32 bits are the LDS byte address, which is what VDST expects.
__device__ __forceinline__ void async_copy_b128(void* lds_dst, const void* gsrc) {
  asm volatile("global_load_async_to_lds_b128 %0, %1, off"
               :: "v"((unsigned)(uintptr_t)lds_dst),
                  "v"((unsigned long long)(uintptr_t)gsrc)
               : "memory");
}
__device__ __forceinline__ void wait_async() {
  asm volatile("s_wait_asynccnt 0" ::: "memory");
}

// K index inside a 16x32 bf16 fragment (ISA 7.12.2): lanes 0-15 hold K-octets
// {0..7, 16..23}, lanes 16-31 hold {8..15, 24..31}, 2 elems per VGPR.
__device__ __forceinline__ int frag_k(int half, int j) {
  return (half << 3) + (j & 7) + ((j & 8) << 1);
}

// A/B fragment from an LDS tile stored row-major [row][k] (ld in elements).
__device__ __forceinline__ v16bf frag_ld_rowmajor(const bf16_t* lds, int ld,
                                                  int row0, int k0) {
  const int l = lane_id();
  const int m = l & 15, half = l >> 4;
  v16bf f;
#pragma unroll
  for (int j = 0; j < 16; ++j)
    f[j] = lds[(row0 + m) * ld + k0 + frag_k(half, j)];
  return f;
}

// Fragment when LDS holds the transpose: lds[k][row].
__device__ __forceinline__ v16bf frag_ld_transposed(const bf16_t* lds, int ld,
                                                    int row0, int k0) {
  const int l = lane_id();
  const int m = l & 15, half = l >> 4;
  v16bf f;
#pragma unroll
  for (int j = 0; j < 16; ++j)
    f[j] = lds[(k0 + frag_k(half, j)) * ld + row0 + m];
  return f;
}

__device__ __forceinline__ v8f wmma_bf16(v16bf a, v16bf b, v8f c) {
  return __builtin_amdgcn_wmma_f32_16x16x32_bf16(false, a, false, b,
                                                 (short)0, c, false, false);
}

// ---------------------------------------------------------------------------
// Kernel 1: QKV = x @ Wqkv^T, fused relu+scale on q,k columns, bf16 out.
// Grid (MROWS/128, QKVN/128), 256 threads (8 waves), wave tile 32x64.
// f32 sources are converted in flight, so staging uses the VGPR path here.
// ---------------------------------------------------------------------------
__global__ void __launch_bounds__(256)
qkv_gemm_kernel(const float* __restrict__ x, const float* __restrict__ Wqkv,
                bf16_t* __restrict__ qkvb) {
  __shared__ bf16_t As[128 * 40];
  __shared__ bf16_t Bs[128 * 40];
  const int m0  = blockIdx.x * 128;
  const int n0  = blockIdx.y * 128;
  const int tid = threadIdx.x;
  const int w   = tid >> 5;
  const int wm  = (w & 3) * 32;
  const int wn  = (w >> 2) * 64;

  v8f acc[2][4];
#pragma unroll
  for (int i = 0; i < 2; ++i)
#pragma unroll
    for (int j = 0; j < 4; ++j) acc[i][j] = (v8f){};

  for (int kk = 0; kk < CDIM; kk += 32) {
    if (kk + 32 < CDIM) {   // L2 prefetch of next K slab (global_prefetch_b8)
      __builtin_prefetch(&x[(size_t)(m0 + (tid & 127)) * CDIM + kk + 32], 0, 1);
      __builtin_prefetch(&Wqkv[(size_t)(n0 + (tid & 127)) * CDIM + kk + 32], 0, 1);
    }
#pragma unroll
    for (int i = 0; i < 4; ++i) {           // A: 128x32 f32 -> bf16 LDS
      int idx = tid + i * 256;
      int r = idx >> 3, c4 = (idx & 7) * 4;
      const float4 v = *(const float4*)&x[(size_t)(m0 + r) * CDIM + kk + c4];
      bf16_t* dst = &As[r * 40 + c4];
      dst[0] = (bf16_t)v.x; dst[1] = (bf16_t)v.y;
      dst[2] = (bf16_t)v.z; dst[3] = (bf16_t)v.w;
    }
#pragma unroll
    for (int i = 0; i < 4; ++i) {           // B: Wqkv rows (out features)
      int idx = tid + i * 256;
      int r = idx >> 3, c4 = (idx & 7) * 4;
      const float4 v = *(const float4*)&Wqkv[(size_t)(n0 + r) * CDIM + kk + c4];
      bf16_t* dst = &Bs[r * 40 + c4];
      dst[0] = (bf16_t)v.x; dst[1] = (bf16_t)v.y;
      dst[2] = (bf16_t)v.z; dst[3] = (bf16_t)v.w;
    }
    __syncthreads();
    v16bf afrag[2], bfrag[4];
#pragma unroll
    for (int i = 0; i < 2; ++i) afrag[i] = frag_ld_rowmajor(As, 40, wm + 16 * i, 0);
#pragma unroll
    for (int j = 0; j < 4; ++j) bfrag[j] = frag_ld_rowmajor(Bs, 40, wn + 16 * j, 0);
#pragma unroll
    for (int i = 0; i < 2; ++i)
#pragma unroll
      for (int j = 0; j < 4; ++j)
        acc[i][j] = wmma_bf16(afrag[i], bfrag[j], acc[i][j]);
    __syncthreads();
  }

  const float qk_scale = 0.125f;  // E^-0.5
  const int l = lane_id();
  const int nlo = l & 15, half = l >> 4;
#pragma unroll
  for (int i = 0; i < 2; ++i)
#pragma unroll
    for (int j = 0; j < 4; ++j) {
      const int col = n0 + wn + 16 * j + nlo;
      const bool is_qk = (col < 2 * CDIM);
#pragma unroll
      for (int r = 0; r < 8; ++r) {
        const int row = m0 + wm + 16 * i + half * 8 + r;
        float vv = acc[i][j][r];
        if (is_qk) vv = fmaxf(vv, 0.f) + qk_scale;
        qkvb[(size_t)row * QKVN + col] = (bf16_t)vv;
      }
    }
}

// ---------------------------------------------------------------------------
// Kernel 2: zero-init atomic accumulators.
// ---------------------------------------------------------------------------
__global__ void zero_kernel(float* __restrict__ p, int n) {
  int i = blockIdx.x * blockDim.x + threadIdx.x;
  if (i < n) p[i] = 0.f;
}

// ---------------------------------------------------------------------------
// Kernel 3: kv[b,h] += k_^T @ v over an N chunk; k_sum via VALU + atomics.
// Grid (B*H, NSEQ/512), 256 threads (8 waves). Staging via async-to-LDS:
// 32x64 bf16 tile = 256 x 16B, exactly one B128 per thread per matrix.
// ---------------------------------------------------------------------------
__global__ void __launch_bounds__(256)
kv_accum_kernel(const bf16_t* __restrict__ qkvb, float* __restrict__ kv,
                float* __restrict__ ksum) {
  __shared__ bf16_t Ks[32 * 72];   // [n][e], 144B pitch (16B-aligned rows)
  __shared__ bf16_t Vs[32 * 72];   // [n][f]
  __shared__ float red[256];
  const int bh = blockIdx.x;
  const int b  = bh >> 3, h = bh & 7;
  const int n0 = blockIdx.y * 512;
  const int tid = threadIdx.x;
  const int w   = tid >> 5;
  const int te  = (w & 3) * 16;
  const int tf0 = (w >> 2) * 32;
  const size_t rowbase = (size_t)b * NSEQ * QKVN;
  const int kcol = CDIM + h * EDIM;
  const int vcol = 2 * CDIM + h * EDIM;

  v8f acc[2];
  acc[0] = (v8f){}; acc[1] = (v8f){};
  float ks_part = 0.f;
  const int scol = tid & 63, srow0 = (tid >> 6) * 8;
  const int cr = tid >> 3, cc8 = (tid & 7) * 8;   // async copy coords

  for (int nc = 0; nc < 512; nc += 32) {
    {
      size_t g = rowbase + (size_t)(n0 + nc + cr) * QKVN;
      async_copy_b128(&Ks[cr * 72 + cc8], &qkvb[g + kcol + cc8]);
      async_copy_b128(&Vs[cr * 72 + cc8], &qkvb[g + vcol + cc8]);
    }
    wait_async();
    __syncthreads();
#pragma unroll
    for (int r = 0; r < 8; ++r)            // k_sum partials
      ks_part += (float)Ks[(srow0 + r) * 72 + scol];
    // A[e][n] = Ks[n][e] ; B[n][f] = Vs[n][f] : both via transposed loader
    v16bf af  = frag_ld_transposed(Ks, 72, te, 0);
    v16bf bf0 = frag_ld_transposed(Vs, 72, tf0, 0);
    v16bf bf1 = frag_ld_transposed(Vs, 72, tf0 + 16, 0);
    acc[0] = wmma_bf16(af, bf0, acc[0]);
    acc[1] = wmma_bf16(af, bf1, acc[1]);
    __syncthreads();
  }

  const int l = lane_id();
  const int nlo = l & 15, half = l >> 4;
#pragma unroll
  for (int j = 0; j < 2; ++j)
#pragma unroll
    for (int r = 0; r < 8; ++r) {
      int e = te + half * 8 + r;
      int f = tf0 + 16 * j + nlo;
      atomicAdd(&kv[((size_t)bh * EDIM + e) * EDIM + f], acc[j][r]);
    }
  red[tid] = ks_part;
  __syncthreads();
  if (tid < 64) {
    float s = red[tid] + red[tid + 64] + red[tid + 128] + red[tid + 192];
    atomicAdd(&ksum[bh * EDIM + tid], s);
  }
}

// ---------------------------------------------------------------------------
// Kernel 4: kv f32 -> bf16.
// ---------------------------------------------------------------------------
__global__ void kv_cvt_kernel(const float* __restrict__ kv,
                              bf16_t* __restrict__ kvb, int n) {
  int i = blockIdx.x * blockDim.x + threadIdx.x;
  if (i < n) kvb[i] = (bf16_t)kv[i];
}

// ---------------------------------------------------------------------------
// Kernel 5: out_attn = (q_ @ kv) * z,  z = 1/(q_ . k_sum + eps).
// Grid (B*H, NSEQ/128), 256 threads. q_/kv tiles staged via async-to-LDS.
// ---------------------------------------------------------------------------
__global__ void __launch_bounds__(256)
attn_out_kernel(const bf16_t* __restrict__ qkvb, const bf16_t* __restrict__ kvb,
                const float* __restrict__ ksum, bf16_t* __restrict__ attn) {
  __shared__ bf16_t Qs[128 * 72];   // [row][e]
  __shared__ bf16_t KVs[64 * 72];   // [e][f]
  __shared__ float ksums[64];
  __shared__ float zs[128];
  const int bh = blockIdx.x;
  const int b  = bh >> 3, h = bh & 7;
  const int n0 = blockIdx.y * 128;
  const int tid = threadIdx.x;
  const int w   = tid >> 5;
  const size_t qbase = (size_t)(b * NSEQ + n0) * QKVN + h * EDIM;

#pragma unroll
  for (int i = 0; i < 2; ++i) {           // kv 64x64 = 512 x 16B
    int idx = tid + i * 256;
    int e = idx >> 3, c8 = (idx & 7) * 8;
    async_copy_b128(&KVs[e * 72 + c8],
                    &kvb[((size_t)bh * EDIM + e) * EDIM + c8]);
  }
#pragma unroll
  for (int i = 0; i < 4; ++i) {           // q_ 128x64 = 1024 x 16B
    int idx = tid + i * 256;
    int r = idx >> 3, c8 = (idx & 7) * 8;
    async_copy_b128(&Qs[r * 72 + c8], &qkvb[qbase + (size_t)r * QKVN + c8]);
  }
  if (tid < 64) ksums[tid] = ksum[bh * EDIM + tid];
  wait_async();
  __syncthreads();
  if (tid < 128) {
    float s = 0.f;
#pragma unroll
    for (int e = 0; e < 64; ++e) s += (float)Qs[tid * 72 + e] * ksums[e];
    zs[tid] = 1.f / (s + 1e-6f);
  }
  __syncthreads();

  v8f acc[4];
#pragma unroll
  for (int j = 0; j < 4; ++j) acc[j] = (v8f){};
#pragma unroll
  for (int ks = 0; ks < 64; ks += 32) {
    v16bf af = frag_ld_rowmajor(Qs, 72, w * 16, ks);
#pragma unroll
    for (int j = 0; j < 4; ++j) {
      v16bf bf = frag_ld_transposed(KVs, 72, j * 16, ks);  // B[k=e][n=f]
      acc[j] = wmma_bf16(af, bf, acc[j]);
    }
  }
  const int l = lane_id();
  const int nlo = l & 15, half = l >> 4;
#pragma unroll
  for (int j = 0; j < 4; ++j)
#pragma unroll
    for (int r = 0; r < 8; ++r) {
      int m = w * 16 + half * 8 + r;
      int f = j * 16 + nlo;
      float vv = acc[j][r] * zs[m];
      attn[(size_t)(b * NSEQ + n0 + m) * CDIM + h * EDIM + f] = (bf16_t)vv;
    }
}

// ---------------------------------------------------------------------------
// Kernel 6: y = out_attn @ Wproj^T + b, f32 output. A-tile staged async.
// ---------------------------------------------------------------------------
__global__ void __launch_bounds__(256)
proj_gemm_kernel(const bf16_t* __restrict__ attn, const float* __restrict__ Wproj,
                 const float* __restrict__ bias, float* __restrict__ out) {
  __shared__ bf16_t As[128 * 40];   // 80B pitch (16B-aligned rows)
  __shared__ bf16_t Bs[128 * 40];
  const int m0  = blockIdx.x * 128;
  const int n0  = blockIdx.y * 128;
  const int tid = threadIdx.x;
  const int w   = tid >> 5;
  const int wm  = (w & 3) * 32;
  const int wn  = (w >> 2) * 64;

  v8f acc[2][4];
#pragma unroll
  for (int i = 0; i < 2; ++i)
#pragma unroll
    for (int j = 0; j < 4; ++j) acc[i][j] = (v8f){};

  for (int kk = 0; kk < CDIM; kk += 32) {
    if (kk + 32 < CDIM)
      __builtin_prefetch(&attn[(size_t)(m0 + (tid & 127)) * CDIM + kk + 32], 0, 1);
#pragma unroll
    for (int i = 0; i < 2; ++i) {          // A bf16 128x32 = 512 x 16B, async
      int idx = tid + i * 256;
      int r = idx >> 2, c8 = (idx & 3) * 8;
      async_copy_b128(&As[r * 40 + c8],
                      &attn[(size_t)(m0 + r) * CDIM + kk + c8]);
    }
#pragma unroll
    for (int i = 0; i < 4; ++i) {          // B: Wproj f32 -> bf16 (VGPR path)
      int idx = tid + i * 256;
      int r = idx >> 3, c4 = (idx & 7) * 4;
      const float4 v = *(const float4*)&Wproj[(size_t)(n0 + r) * CDIM + kk + c4];
      bf16_t* dst = &Bs[r * 40 + c4];
      dst[0] = (bf16_t)v.x; dst[1] = (bf16_t)v.y;
      dst[2] = (bf16_t)v.z; dst[3] = (bf16_t)v.w;
    }
    wait_async();
    __syncthreads();
    v16bf afrag[2], bfrag[4];
#pragma unroll
    for (int i = 0; i < 2; ++i) afrag[i] = frag_ld_rowmajor(As, 40, wm + 16 * i, 0);
#pragma unroll
    for (int j = 0; j < 4; ++j) bfrag[j] = frag_ld_rowmajor(Bs, 40, wn + 16 * j, 0);
#pragma unroll
    for (int i = 0; i < 2; ++i)
#pragma unroll
      for (int j = 0; j < 4; ++j)
        acc[i][j] = wmma_bf16(afrag[i], bfrag[j], acc[i][j]);
    __syncthreads();
  }

  const int l = lane_id();
  const int nlo = l & 15, half = l >> 4;
#pragma unroll
  for (int i = 0; i < 2; ++i)
#pragma unroll
    for (int j = 0; j < 4; ++j) {
      const int col = n0 + wn + 16 * j + nlo;
      const float bv = bias[col];
#pragma unroll
      for (int r = 0; r < 8; ++r) {
        const int row = m0 + wm + 16 * i + half * 8 + r;
        out[(size_t)row * CDIM + col] = acc[i][j][r] + bv;
      }
    }
}

extern "C" void kernel_launch(void* const* d_in, const int* in_sizes, int n_in,
                              void* d_out, int out_size, void* d_ws, size_t ws_size,
                              hipStream_t stream) {
  const float* x     = (const float*)d_in[0];
  const float* Wqkv  = (const float*)d_in[1];
  const float* Wproj = (const float*)d_in[2];
  const float* bproj = (const float*)d_in[3];
  float* out = (float*)d_out;

  char* ws = (char*)d_ws;
  size_t off = 0;
  bf16_t* qkvb = (bf16_t*)(ws + off); off += (size_t)MROWS * QKVN * sizeof(bf16_t);
  float*  kv   = (float*)(ws + off);  off += (size_t)BATCH * HNUM * EDIM * EDIM * sizeof(float);
  float*  ksum = (float*)(ws + off);  off += (size_t)BATCH * HNUM * EDIM * sizeof(float);
  bf16_t* kvb  = (bf16_t*)(ws + off); off += (size_t)BATCH * HNUM * EDIM * EDIM * sizeof(bf16_t);
  bf16_t* attn = (bf16_t*)(ws + off); off += (size_t)MROWS * CDIM * sizeof(bf16_t);

  qkv_gemm_kernel<<<dim3(MROWS / 128, QKVN / 128), 256, 0, stream>>>(x, Wqkv, qkvb);

  const int nz = BATCH * HNUM * EDIM * EDIM + BATCH * HNUM * EDIM; // kv + ksum contiguous
  zero_kernel<<<(nz + 255) / 256, 256, 0, stream>>>(kv, nz);

  kv_accum_kernel<<<dim3(BATCH * HNUM, NSEQ / 512), 256, 0, stream>>>(qkvb, kv, ksum);

  const int nkv = BATCH * HNUM * EDIM * EDIM;
  kv_cvt_kernel<<<(nkv + 255) / 256, 256, 0, stream>>>(kv, kvb, nkv);

  attn_out_kernel<<<dim3(BATCH * HNUM, NSEQ / 128), 256, 0, stream>>>(qkvb, kvb, ksum, attn);

  proj_gemm_kernel<<<dim3(MROWS / 128, CDIM / 128), 256, 0, stream>>>(attn, Wproj, bproj, out);
}
